// MSSP_54520314856180
// MI455X (gfx1250) — matrix operand order repaired and checked by reading, not compiled, not run on hardware
//
#include <hip/hip_runtime.h>

// ---------------------------------------------------------------------------
// Problem constants (from reference): B=256, S=128, P=64, H=768, C=2187
// ---------------------------------------------------------------------------
constexpr int B_ = 256;
constexpr int S_ = 128;
constexpr int P_ = 64;
constexpr int H_ = 768;
constexpr int C_ = 2187;
constexpr float NEGV = -1.0e7f;

constexpr int NT  = 128;       // class-tile (N) per block
constexpr int KC  = 64;        // K chunk staged through LDS
constexpr int LDA = KC + 8;    // LDS row stride in bf16 elems (bank-skewed)
constexpr int NCH = H_ / KC;   // 12 K-chunks

typedef __attribute__((ext_vector_type(16))) __bf16 v16bf;
typedef __attribute__((ext_vector_type(8)))  float  v8f;

__device__ __forceinline__ unsigned short f2bf(float f) {
    unsigned u = __float_as_uint(f);
    unsigned r = (u + 0x7FFFu + ((u >> 16) & 1u)) >> 16;   // RNE
    return (unsigned short)r;
}

#if __has_builtin(__builtin_amdgcn_cvt_pk_bf16_f32)
__device__ __forceinline__ unsigned pk2(float a, float b) {
    auto r = __builtin_amdgcn_cvt_pk_bf16_f32(a, b);       // v_cvt_pk_bf16_f32
    unsigned u;
    __builtin_memcpy(&u, &r, sizeof(u));
    return u;
}
#else
__device__ __forceinline__ unsigned pk2(float a, float b) {
    return (unsigned)f2bf(a) | ((unsigned)f2bf(b) << 16);
}
#endif

__device__ __forceinline__ uint2 pack4(float4 f) {
    return make_uint2(pk2(f.x, f.y), pk2(f.z, f.w));
}

// Async global->LDS copy (CDNA5 ASYNCcnt path), 16B per lane.  Inline asm is
// used because the clang builtin arity differs across toolchains.
__device__ __forceinline__ void async_b128(unsigned ldsByteOff, unsigned long long gaddr) {
    asm volatile("global_load_async_to_lds_b128 %0, %1, off"
                 :: "v"(ldsByteOff), "v"(gaddr) : "memory");
}
__device__ __forceinline__ void wait_async0() {
    asm volatile("s_wait_asynccnt 0x0" ::: "memory");
}

// ---------------------------------------------------------------------------
// Pre-pass 1: W f32 -> bf16 (done once instead of by all 256 batch blocks)
// ---------------------------------------------------------------------------
__global__ __launch_bounds__(256) void convert_w_kernel(
    const float* __restrict__ W, unsigned short* __restrict__ out)
{
    const int n4 = C_ * H_ / 4;
    for (int idx = blockIdx.x * blockDim.x + threadIdx.x; idx < n4;
         idx += gridDim.x * blockDim.x) {
        ((uint2*)out)[idx] = pack4(((const float4*)W)[idx]);
    }
}

// ---------------------------------------------------------------------------
// Pre-pass 2: gather piece rows h[i, index[i,p], :] -> bf16
// grid = B*P blocks, 192 threads (x4 floats = 768 = H)
// ---------------------------------------------------------------------------
__global__ __launch_bounds__(192) void gather_convert_kernel(
    const float* __restrict__ h, const long long* __restrict__ index,
    unsigned short* __restrict__ out)
{
    const int bp = blockIdx.x;                 // i*P + p
    const int i  = bp / P_;
    const int s  = (int)index[bp];
    const float4* src = (const float4*)(h + ((size_t)i * S_ + s) * H_);
    uint2* dst = (uint2*)(out + (size_t)bp * H_);
    dst[threadIdx.x] = pack4(src[threadIdx.x]);
}

// ---------------------------------------------------------------------------
// Kernel: fused (gather+) GEMM + bias + mask + max over P  -> score
// grid = (ceil(C/128), B), block = 256 (8 waves), M=64 N=128 K=768
// Double-buffered LDS pipeline.  Two staging paths:
//   useAsync=1: bf16 tiles copied global->LDS via global_load_async_to_lds_b128
//   useAsync=0: f32 loads -> v_cvt_pk_bf16_f32 -> ds_store (register pipeline)
// ---------------------------------------------------------------------------
__global__ __launch_bounds__(256) void fused_gemm_max_kernel(
    const float* __restrict__ h,          // [B,S,H]
    const long long* __restrict__ index,  // [B,P] int64
    const int* __restrict__ imask,        // [B,P]
    const float* __restrict__ W,          // [C,H] f32
    const float* __restrict__ bias,       // [C]
    float* __restrict__ score,            // [B,C]
    const unsigned short* __restrict__ Abf, // [B*P,H] bf16 (ws, useAsync only)
    const unsigned short* __restrict__ Wbf, // [C,H]   bf16 (ws, useAsync only)
    int useAsync)
{
    __shared__ __align__(16) unsigned short sA[2][P_ * LDA];
    __shared__ __align__(16) unsigned short sW[2][NT * LDA];
    __shared__ float sMaxArr[2][NT];
    __shared__ int sIdx[P_];
    __shared__ int sMsk[P_];

    const int tid = threadIdx.x;
    const int i   = blockIdx.y;
    const int c0  = blockIdx.x * NT;

    if (tid < P_) {
        sIdx[tid] = (int)index[(size_t)i * P_ + tid];
        sMsk[tid] = imask[(size_t)i * P_ + tid];
    }
    __syncthreads();

    const int lane = tid & 31;
    const int wave = tid >> 5;
    const int wm   = wave & 1;     // M group: rows wm*32 .. wm*32+31
    const int wn   = wave >> 1;    // N group: cols wn*32 .. wn*32+31
    const int lm   = lane & 15;
    const int lh   = lane >> 4;

    v8f acc[2][2] = {};

    auto compute = [&](int cur) {
        #pragma unroll
        for (int ks = 0; ks < KC; ks += 32) {
            union FU { v16bf v; unsigned u[8]; } a0, a1, b0, b1;
            #pragma unroll
            for (int p = 0; p < 8; ++p) {
                // A 16-bit 16x32: lanes 0-15 K={0..7,16..23}, lanes 16-31 +8/+24
                const int grp = p >> 2, w = p & 3;
                const int kkA = ks + grp * 16 + lh * 8 + w * 2;
                a0.u[p] = *(const unsigned*)&sA[cur][(wm * 32 +      lm) * LDA + kkA];
                a1.u[p] = *(const unsigned*)&sA[cur][(wm * 32 + 16 + lm) * LDA + kkA];
                // B 32x16: lanes 0-15 K={0..7,16..23}, lanes 16-31 +8/+24
                const int kkB = ks + grp * 16 + lh * 8 + w * 2;
                b0.u[p] = *(const unsigned*)&sW[cur][(wn * 32 +      lm) * LDA + kkB];
                b1.u[p] = *(const unsigned*)&sW[cur][(wn * 32 + 16 + lm) * LDA + kkB];
            }
            acc[0][0] = __builtin_amdgcn_wmma_f32_16x16x32_bf16(
                false, a0.v, false, b0.v, (short)0, acc[0][0], false, false);
            acc[0][1] = __builtin_amdgcn_wmma_f32_16x16x32_bf16(
                false, a0.v, false, b1.v, (short)0, acc[0][1], false, false);
            acc[1][0] = __builtin_amdgcn_wmma_f32_16x16x32_bf16(
                false, a1.v, false, b0.v, (short)0, acc[1][0], false, false);
            acc[1][1] = __builtin_amdgcn_wmma_f32_16x16x32_bf16(
                false, a1.v, false, b1.v, (short)0, acc[1][1], false, false);
        }
    };

    if (useAsync) {
        // ---- async bf16 staging: 16B chunks, A: 512, W: 1024 per K-chunk ----
        auto stageAsync = [&](int k0, int buf) {
            #pragma unroll
            for (int j = 0; j < 2; ++j) {                  // A tile
                const int id  = tid + j * 256;
                const int row = id >> 3, seg = id & 7;
                unsigned lds = (unsigned)(size_t)&sA[buf][row * LDA + seg * 8];
                unsigned long long ga = (unsigned long long)
                    (Abf + (size_t)(i * P_ + row) * H_ + k0 + seg * 8);
                async_b128(lds, ga);
            }
            #pragma unroll
            for (int j = 0; j < 4; ++j) {                  // W tile
                const int id  = tid + j * 256;
                const int row = id >> 3, seg = id & 7;
                const int cr  = (c0 + row < C_) ? (c0 + row) : (C_ - 1); // clamp
                unsigned lds = (unsigned)(size_t)&sW[buf][row * LDA + seg * 8];
                unsigned long long ga = (unsigned long long)
                    (Wbf + (size_t)cr * H_ + k0 + seg * 8);
                async_b128(lds, ga);
            }
        };
        stageAsync(0, 0);
        #pragma unroll 1
        for (int kc = 0; kc < NCH; ++kc) {
            wait_async0();                 // our async stores for chunk kc landed
            __syncthreads();               // everyone's landed
            if (kc + 1 < NCH) stageAsync((kc + 1) * KC, (kc + 1) & 1);
            compute(kc & 1);               // overlaps next chunk's DMA
        }
    } else {
        // ---- register pipeline: f32 loads -> cvt_pk_bf16 -> ds_store ----
        const int arow = tid >> 2, aq = tid & 3;   // A: 64 rows x 4 thr, 16 f
        const int wrow = tid >> 1, wq = tid & 1;   // W: 128 rows x 2 thr, 32 f
        const bool wok = (c0 + wrow) < C_;
        const int rowIdx = sIdx[arow];
        const float* hRow = h + ((size_t)i * S_ + rowIdx) * H_ + aq * 16;
        const float* wRow = W + (size_t)(wok ? c0 + wrow : 0) * H_ + wq * 32;

        float4 aR[4], wR[8];
        auto loadRegs = [&](int k0) {
            const float4* sa = (const float4*)(hRow + k0);
            #pragma unroll
            for (int v = 0; v < 4; ++v) aR[v] = sa[v];
            if (wok) {
                const float4* sw = (const float4*)(wRow + k0);
                #pragma unroll
                for (int v = 0; v < 8; ++v) wR[v] = sw[v];
                if (k0 + KC < H_) __builtin_prefetch(wRow + k0 + KC, 0, 1);
            } else {
                #pragma unroll
                for (int v = 0; v < 8; ++v) wR[v] = make_float4(0, 0, 0, 0);
            }
        };
        auto storeRegs = [&](int buf) {
            uint2* dA = (uint2*)&sA[buf][arow * LDA + aq * 16];
            #pragma unroll
            for (int v = 0; v < 4; ++v) dA[v] = pack4(aR[v]);
            uint2* dW = (uint2*)&sW[buf][wrow * LDA + wq * 32];
            #pragma unroll
            for (int v = 0; v < 8; ++v) dW[v] = pack4(wR[v]);
        };

        loadRegs(0);
        #pragma unroll 1
        for (int kc = 0; kc < NCH; ++kc) {
            storeRegs(kc & 1);
            if (kc + 1 < NCH) loadRegs((kc + 1) * KC);  // hidden behind compute
            __syncthreads();
            compute(kc & 1);
        }
    }

    // ---- epilogue: + bias, mask -> -1e7, max over P ----
    #pragma unroll
    for (int nt = 0; nt < 2; ++nt) {
        const int cLoc = wn * 32 + nt * 16 + lm;
        const int c    = c0 + cLoc;
        const float bv = (c < C_) ? bias[c] : 0.0f;
        float best = -3.0e38f;
        #pragma unroll
        for (int mt = 0; mt < 2; ++mt) {
            #pragma unroll
            for (int r = 0; r < 8; ++r) {
                // C/D layout: lanes 0-15 hold M=r, lanes 16-31 hold M=r+8
                const int p = wm * 32 + mt * 16 + lh * 8 + r;
                const float v = sMsk[p] ? (acc[mt][nt][r] + bv) : NEGV;
                best = fmaxf(best, v);
            }
        }
        best = fmaxf(best, __shfl_xor(best, 16, 32));   // merge lane halves
        if (lh == 0) sMaxArr[wm][cLoc] = best;
    }
    __syncthreads();
    if (tid < NT) {
        const int c = c0 + tid;
        if (c < C_) {
            score[(size_t)i * C_ + c] = fmaxf(sMaxArr[0][tid], sMaxArr[1][tid]);
        }
    }
}

// ---------------------------------------------------------------------------
// Loss: -mean(y*log_sigmoid(s) + (1-y)*log_sigmoid(-s))
// ---------------------------------------------------------------------------
__global__ __launch_bounds__(256) void loss_kernel(
    const float* __restrict__ score, const float* __restrict__ y,
    float* __restrict__ loss)
{
    __shared__ float red[256];
    const int n = B_ * C_;
    const float invN = -1.0f / (float)n;
    float acc = 0.0f;
    for (int idx = blockIdx.x * blockDim.x + threadIdx.x; idx < n;
         idx += gridDim.x * blockDim.x) {
        const float s  = score[idx];
        const float yy = y[idx];
        const float l1 = log1pf(__expf(-fabsf(s)));
        acc += yy * (fminf(s, 0.0f) - l1) + (1.0f - yy) * (fminf(-s, 0.0f) - l1);
    }
    red[threadIdx.x] = acc * invN;
    __syncthreads();
    for (int off = 128; off > 0; off >>= 1) {
        if (threadIdx.x < off) red[threadIdx.x] += red[threadIdx.x + off];
        __syncthreads();
    }
    if (threadIdx.x == 0) atomicAdd(loss, red[0]);
}

// ---------------------------------------------------------------------------
// Per-row descending stable argsort via rank scatter (matches argsort(-score))
// ---------------------------------------------------------------------------
__global__ __launch_bounds__(256) void rank_kernel(
    const float* __restrict__ score, int* __restrict__ indices)
{
    __shared__ float srow[C_];
    const int i = blockIdx.x;
    for (int c = threadIdx.x; c < C_; c += blockDim.x)
        srow[c] = score[(size_t)i * C_ + c];
    __syncthreads();
    for (int c = threadIdx.x; c < C_; c += blockDim.x) {
        const float sc = srow[c];
        int rank = 0;
        for (int j = 0; j < C_; ++j) {
            const float sj = srow[j];
            rank += (sj > sc) || ((sj == sc) && (j < c));
        }
        indices[(size_t)i * C_ + rank] = c;
    }
}

// ---------------------------------------------------------------------------
// Launch: d_in = {h, y, index(i64), index_mask(i32), W, b}
// d_out = [loss(1) | score(B*C) f32 | indices(B*C) i32] flat
// ---------------------------------------------------------------------------
extern "C" void kernel_launch(void* const* d_in, const int* in_sizes, int n_in,
                              void* d_out, int out_size, void* d_ws, size_t ws_size,
                              hipStream_t stream) {
    (void)in_sizes; (void)n_in; (void)out_size;
    const float*     h     = (const float*)d_in[0];
    const float*     y     = (const float*)d_in[1];
    const long long* index = (const long long*)d_in[2];
    const int*       imask = (const int*)d_in[3];
    const float*     W     = (const float*)d_in[4];
    const float*     bias  = (const float*)d_in[5];

    float* loss    = (float*)d_out;
    float* score   = loss + 1;
    int*   indices = (int*)(score + (size_t)B_ * C_);

    // Workspace plan: Wbf [C*H] bf16, Abf [B*P*H] bf16
    const size_t needW = (size_t)C_ * H_ * sizeof(unsigned short);
    const size_t needA = (size_t)B_ * P_ * H_ * sizeof(unsigned short);
    const int useAsync = (ws_size >= needW + needA) ? 1 : 0;
    unsigned short* Wbf = (unsigned short*)d_ws;
    unsigned short* Abf = Wbf + (size_t)C_ * H_;

    hipMemsetAsync(loss, 0, sizeof(float), stream);   // capturable

    if (useAsync) {
        convert_w_kernel<<<512, 256, 0, stream>>>(W, Wbf);
        gather_convert_kernel<<<B_ * P_, 192, 0, stream>>>(h, index, Abf);
    }
    dim3 grid((C_ + NT - 1) / NT, B_);
    fused_gemm_max_kernel<<<grid, 256, 0, stream>>>(h, index, imask, W, bias,
                                                    score, Abf, Wbf, useAsync);
    loss_kernel<<<512, 256, 0, stream>>>(score, y, loss);
    rank_kernel<<<B_, 256, 0, stream>>>(score, indices);
}
